// TransformerEncoderLayer_5403068859005
// MI455X (gfx1250) — matrix operand order repaired and checked by reading, not compiled, not run on hardware
//
#include <hip/hip_runtime.h>
#include <hip/hip_bf16.h>
#include <cstdint>

typedef unsigned short u16;
typedef __attribute__((ext_vector_type(16))) __bf16 v16bf;
typedef __attribute__((ext_vector_type(8)))  float  v8f;

// ---------- helpers ----------
static __device__ __forceinline__ u16 f2bf(float f) {
    union { float f; uint32_t u; } c; c.f = f;
    uint32_t u = c.u;
    u += 0x7fffu + ((u >> 16) & 1u);   // round-to-nearest-even
    return (u16)(u >> 16);
}

static __device__ __forceinline__ v8f wmma_bf16(v16bf a, v16bf b, v8f c) {
    return __builtin_amdgcn_wmma_f32_16x16x32_bf16(false, a, false, b, (short)0, c, false, false);
}

// CDNA5 async global->LDS copy (ASYNCcnt-tracked, bypasses VGPRs)
static __device__ __forceinline__ void async_copy_b128(void* lds_ptr, const void* gptr) {
    uint32_t lds_addr = (uint32_t)(uintptr_t)lds_ptr;          // low 32 bits of generic = LDS offset
    uint64_t gaddr    = (uint64_t)(uintptr_t)gptr;
    asm volatile("global_load_async_to_lds_b128 %0, %1, off"
                 :: "v"(lds_addr), "v"(gaddr) : "memory");
}

static __device__ __forceinline__ void wait_async_le4() {
#if __has_builtin(__builtin_amdgcn_s_wait_asynccnt)
    __builtin_amdgcn_s_wait_asynccnt(4);
#else
    asm volatile("s_wait_asynccnt 0x4" ::: "memory");
#endif
}
static __device__ __forceinline__ void wait_async_0() {
#if __has_builtin(__builtin_amdgcn_s_wait_asynccnt)
    __builtin_amdgcn_s_wait_asynccnt(0);
#else
    asm volatile("s_wait_asynccnt 0x0" ::: "memory");
#endif
}

// A-operand fragment: 16 rows (M = lane&15), K chunks {0-7,16-23} (lo half) / {8-15,24-31} (hi half)
static __device__ __forceinline__ v16bf ld_fragA(const u16* base, int rowStride, int rowBase,
                                                 int kBase, int lane) {
    const u16* p = base + (size_t)(rowBase + (lane & 15)) * rowStride + kBase + ((lane >> 4) << 3);
    union { uint4 q[2]; v16bf v; } u;
    u.q[0] = *(const uint4*)p;
    u.q[1] = *(const uint4*)(p + 16);
    return u.v;
}

// B-operand fragment (Bt stored N-major, K contiguous): N = lane&15, K chunks {0-7,16-23} (lo half) / {8-15,24-31} (hi half)
static __device__ __forceinline__ v16bf ld_fragB(const u16* base, int rowStride, int rowBase,
                                                 int kBase, int lane) {
    const u16* p = base + (size_t)(rowBase + (lane & 15)) * rowStride + kBase + ((lane >> 4) << 3);
    union { uint4 q[2]; v16bf v; } u;
    u.q[0] = *(const uint4*)p;
    u.q[1] = *(const uint4*)(p + 16);
    return u.v;
}

static __device__ __forceinline__ float gelu_exact(float v) {
    return 0.5f * v * (1.0f + erff(v * 0.70710678118654752f));
}

// ---------- weight fp32 [K][N] -> bf16 [N][K] ----------
__global__ void convT_kernel(const float* __restrict__ W, u16* __restrict__ Wt,
                             int K, int N) {
    uint32_t idx = blockIdx.x * 256u + threadIdx.x;   // total K*N, multiple of 256
    uint32_t k = idx / (uint32_t)N;
    uint32_t n = idx - k * (uint32_t)N;
    Wt[(size_t)n * K + k] = f2bf(W[idx]);
}

// ---------- LayerNorm over D=1024, optional residual, f32 and/or bf16 out ----------
__global__ __launch_bounds__(256) void ln_kernel(const float* __restrict__ X,
                                                 const float* __restrict__ R,
                                                 const float* __restrict__ g,
                                                 const float* __restrict__ bta,
                                                 float* __restrict__ outF,
                                                 u16* __restrict__ outB) {
    const int D = 1024;
    int row = blockIdx.x, tid = threadIdx.x;
    const float* xr = X + (size_t)row * D;
    const float* rr = R ? R + (size_t)row * D : nullptr;
    float v[4];
    float s1 = 0.f, s2 = 0.f;
#pragma unroll
    for (int i = 0; i < 4; ++i) {
        int c = tid + i * 256;
        float t = xr[c];
        if (rr) t += rr[c];
        v[i] = t; s1 += t; s2 += t * t;
    }
#pragma unroll
    for (int off = 1; off < 32; off <<= 1) {
        s1 += __shfl_xor(s1, off, 32);
        s2 += __shfl_xor(s2, off, 32);
    }
    __shared__ float a1[8], a2[8];
    int w = tid >> 5, ln = tid & 31;
    if (ln == 0) { a1[w] = s1; a2[w] = s2; }
    __syncthreads();
    float t1 = 0.f, t2 = 0.f;
#pragma unroll
    for (int j = 0; j < 8; ++j) { t1 += a1[j]; t2 += a2[j]; }
    float mu = t1 * (1.0f / D);
    float var = t2 * (1.0f / D) - mu * mu;
    float rs = rsqrtf(var + 1e-5f);
#pragma unroll
    for (int i = 0; i < 4; ++i) {
        int c = tid + i * 256;
        float o = (v[i] - mu) * rs * g[c] + bta[c];
        if (outF) outF[(size_t)row * D + c] = o;
        if (outB) outB[(size_t)row * D + c] = f2bf(o);
    }
}

// ---------- bf16 WMMA GEMM: C[M,N] = A[M,K] * Bt[N,K]^T + bias (+gelu) (+residual) ----------
// Double-buffered LDS tiles fed by async global->LDS DMA overlapped with WMMA.
__global__ __launch_bounds__(256) void gemm_bf16_kernel(const u16* __restrict__ A,
                                                        const u16* __restrict__ Bt,
                                                        const float* __restrict__ bias,
                                                        const float* __restrict__ residual,
                                                        float* __restrict__ outF,
                                                        u16* __restrict__ outB,
                                                        int M, int N, int K, int act_gelu) {
    constexpr int LDT = 40;                 // padded halfword stride -> conflict-free frag reads
    __shared__ __align__(16) u16 As[2][128 * LDT];
    __shared__ __align__(16) u16 Bs[2][128 * LDT];
    int tid = threadIdx.x, lane = tid & 31, wave = tid >> 5;
    int mBase = blockIdx.y * 128, nBase = blockIdx.x * 128;
    int rowW = (wave & 3) * 32;             // 4 wave rows * 32
    int colW = (wave >> 2) * 64;            // 2 wave cols * 64
    v8f acc[2][4];
    v8f z = {0.f, 0.f, 0.f, 0.f, 0.f, 0.f, 0.f, 0.f};
#pragma unroll
    for (int i = 0; i < 2; ++i)
#pragma unroll
        for (int j = 0; j < 4; ++j) acc[i][j] = z;

    // per-thread staging slots: 512 b128 chunks per (A|B) tile, 2 per thread each
    int r0 = tid >> 2,            kc0 = (tid & 3) * 8;
    int r1 = (tid + 256) >> 2,    kc1 = ((tid + 256) & 3) * 8;

    auto issue_tile = [&](int k0, int buf) {
        async_copy_b128(&As[buf][r0 * LDT + kc0], A  + (size_t)(mBase + r0) * K + k0 + kc0);
        async_copy_b128(&As[buf][r1 * LDT + kc1], A  + (size_t)(mBase + r1) * K + k0 + kc1);
        async_copy_b128(&Bs[buf][r0 * LDT + kc0], Bt + (size_t)(nBase + r0) * K + k0 + kc0);
        async_copy_b128(&Bs[buf][r1 * LDT + kc1], Bt + (size_t)(nBase + r1) * K + k0 + kc1);
    };

    issue_tile(0, 0);                       // prologue: tile 0 in flight
    const int T = K / 32;
    for (int it = 0; it < T; ++it) {
        int buf = it & 1;
        __syncthreads();                    // all waves done reading buf^1 (tile it-1)
        if (it + 1 < T) issue_tile((it + 1) * 32, buf ^ 1);
        if (it + 1 < T) wait_async_le4();   // tile it landed (in-order), tile it+1 still in flight
        else            wait_async_0();
        __syncthreads();                    // all waves' async data for tile it visible
        v16bf a0 = ld_fragA(As[buf], LDT, rowW + 0, 0, lane);
        v16bf a1 = ld_fragA(As[buf], LDT, rowW + 16, 0, lane);
#pragma unroll
        for (int ct = 0; ct < 4; ++ct) {
            v16bf b = ld_fragB(Bs[buf], LDT, colW + ct * 16, 0, lane);
            acc[0][ct] = wmma_bf16(a0, b, acc[0][ct]);
            acc[1][ct] = wmma_bf16(a1, b, acc[1][ct]);
        }
    }

#pragma unroll
    for (int rt = 0; rt < 2; ++rt)
#pragma unroll
        for (int ct = 0; ct < 4; ++ct) {
            int col = nBase + colW + ct * 16 + (lane & 15);
            float bc = bias ? bias[col] : 0.f;
#pragma unroll
            for (int r = 0; r < 8; ++r) {
                int row = mBase + rowW + rt * 16 + r + ((lane >> 4) << 3);
                float vv = acc[rt][ct][r] + bc;
                if (act_gelu) vv = gelu_exact(vv);
                if (residual) vv += residual[(size_t)row * N + col];
                if (outF) outF[(size_t)row * N + col] = vv;
                if (outB) outB[(size_t)row * N + col] = f2bf(vv);
            }
        }
}

// ---------- flash attention: qkv bf16 [B*S, 3*1024] -> ctx bf16 [B*S, 1024] ----------
__global__ __launch_bounds__(256) void flash_attn_kernel(const u16* __restrict__ qkv,
                                                         u16* __restrict__ ctx) {
    constexpr int S = 2048, D3 = 3072, HD = 64;
    __shared__ __align__(16) u16 Ks[128 * 72];      // [kpos][d], padded
    __shared__ __align__(16) u16 Vt[64 * 136];      // [d][kpos], padded
    __shared__ __align__(16) u16 Ps[8 * 16 * 136];  // per-wave P tile [16][128], padded
    int tid = threadIdx.x, lane = tid & 31, wave = tid >> 5;
    int b = blockIdx.z, h = blockIdx.y, qt = blockIdx.x;
    int rowQ = qt * 128 + wave * 16;                // q-row base within batch for this wave

    // Q fragments live in registers for the whole kernel
    const u16* qbase = qkv + (size_t)(b * S + rowQ + (lane & 15)) * D3 + h * HD;
    v16bf qf[2];
#pragma unroll
    for (int ds2 = 0; ds2 < 2; ++ds2) {
        int k = ds2 * 32 + ((lane >> 4) << 3);
        union { uint4 q[2]; v16bf v; } u;
        u.q[0] = *(const uint4*)(qbase + k);
        u.q[1] = *(const uint4*)(qbase + k + 16);
        qf[ds2] = u.v;
    }

    float m[8], l[8];
    v8f o[4];
    v8f z = {0.f, 0.f, 0.f, 0.f, 0.f, 0.f, 0.f, 0.f};
#pragma unroll
    for (int r = 0; r < 8; ++r) { m[r] = -1e30f; l[r] = 0.f; }
#pragma unroll
    for (int dj = 0; dj < 4; ++dj) o[dj] = z;

    u16* Pw = Ps + wave * 16 * 136;

    for (int kt = 0; kt < 16; ++kt) {
        __syncthreads();                            // prior tile fully consumed
        const u16* kg = qkv + (size_t)(b * S + kt * 128) * D3 + 1024 + h * HD;
        const u16* vg = qkv + (size_t)(b * S + kt * 128) * D3 + 2048 + h * HD;
        // K tile: async DMA straight into LDS
#pragma unroll
        for (int i = 0; i < 4; ++i) {
            int v = tid + i * 256;                  // 1024 b128 chunks
            int r = v >> 3, dc = (v & 7) * 8;
            async_copy_b128(Ks + r * 72 + dc, kg + (size_t)r * D3 + dc);
        }
        // V tile: batch all global loads first, then transpose-scatter to LDS
        union { uint4 q; u16 h8[8]; } vv[4];
        int vr[4], vdc[4];
#pragma unroll
        for (int i = 0; i < 4; ++i) {
            int v = tid + i * 256;
            vr[i] = v >> 3; vdc[i] = (v & 7) * 8;
            vv[i].q = *(const uint4*)(vg + (size_t)vr[i] * D3 + vdc[i]);
        }
#pragma unroll
        for (int i = 0; i < 4; ++i)
#pragma unroll
            for (int j = 0; j < 8; ++j) Vt[(vdc[i] + j) * 136 + vr[i]] = vv[i].h8[j];
        wait_async_0();
        __syncthreads();

        // S = Q * K^T  (scaled)
        v8f s[8];
#pragma unroll
        for (int kj = 0; kj < 8; ++kj) s[kj] = z;
#pragma unroll
        for (int ds2 = 0; ds2 < 2; ++ds2)
#pragma unroll
            for (int kj = 0; kj < 8; ++kj)
                s[kj] = wmma_bf16(qf[ds2], ld_fragB(Ks, 72, kj * 16, ds2 * 32, lane), s[kj]);

        float rmax[8];
#pragma unroll
        for (int r = 0; r < 8; ++r) rmax[r] = -1e30f;
#pragma unroll
        for (int kj = 0; kj < 8; ++kj)
#pragma unroll
            for (int r = 0; r < 8; ++r) {
                s[kj][r] *= 0.125f;                 // 1/sqrt(64)
                rmax[r] = fmaxf(rmax[r], s[kj][r]);
            }
#pragma unroll
        for (int r = 0; r < 8; ++r)
#pragma unroll
            for (int off = 1; off < 16; off <<= 1)
                rmax[r] = fmaxf(rmax[r], __shfl_xor(rmax[r], off, 32));

        float alpha[8], rsum[8];
#pragma unroll
        for (int r = 0; r < 8; ++r) {
            float mn = fmaxf(m[r], rmax[r]);
            alpha[r] = __expf(m[r] - mn);
            m[r] = mn;
            rsum[r] = 0.f;
        }
        // P = exp(S - m); stage to per-wave LDS for C-layout -> A-layout conversion
#pragma unroll
        for (int kj = 0; kj < 8; ++kj)
#pragma unroll
            for (int r = 0; r < 8; ++r) {
                float p = __expf(s[kj][r] - m[r]);
                rsum[r] += p;
                Pw[(r + ((lane >> 4) << 3)) * 136 + kj * 16 + (lane & 15)] = f2bf(p);
            }
#pragma unroll
        for (int r = 0; r < 8; ++r) {
#pragma unroll
            for (int off = 1; off < 16; off <<= 1)
                rsum[r] += __shfl_xor(rsum[r], off, 32);
            l[r] = l[r] * alpha[r] + rsum[r];
        }
#pragma unroll
        for (int dj = 0; dj < 4; ++dj)
#pragma unroll
            for (int r = 0; r < 8; ++r) o[dj][r] *= alpha[r];

        // O += P * V
#pragma unroll
        for (int ks = 0; ks < 4; ++ks) {
            v16bf aP = ld_fragA(Pw, 136, 0, ks * 32, lane);
#pragma unroll
            for (int dj = 0; dj < 4; ++dj)
                o[dj] = wmma_bf16(aP, ld_fragB(Vt, 136, dj * 16, ks * 32, lane), o[dj]);
        }
    }

#pragma unroll
    for (int dj = 0; dj < 4; ++dj)
#pragma unroll
        for (int r = 0; r < 8; ++r) {
            int srow = b * S + rowQ + r + ((lane >> 4) << 3);
            ctx[(size_t)srow * 1024 + h * HD + dj * 16 + (lane & 15)] =
                f2bf(o[dj][r] * (1.0f / l[r]));
        }
}

// ---------- host orchestration ----------
extern "C" void kernel_launch(void* const* d_in, const int* in_sizes, int n_in,
                              void* d_out, int out_size, void* d_ws, size_t ws_size,
                              hipStream_t stream) {
    const float* x        = (const float*)d_in[0];
    const float* qkv_w    = (const float*)d_in[1];
    const float* qkv_b    = (const float*)d_in[2];
    const float* out_w    = (const float*)d_in[3];
    const float* out_b    = (const float*)d_in[4];
    const float* attn_ln_g = (const float*)d_in[5];
    const float* attn_ln_b = (const float*)d_in[6];
    const float* norm1_g  = (const float*)d_in[7];
    const float* norm1_b  = (const float*)d_in[8];
    const float* ff_ln_g  = (const float*)d_in[9];
    const float* ff_ln_b  = (const float*)d_in[10];
    const float* w1       = (const float*)d_in[11];
    const float* b1       = (const float*)d_in[12];
    const float* w2       = (const float*)d_in[13];
    const float* b2       = (const float*)d_in[14];
    const float* norm2_g  = (const float*)d_in[15];
    const float* norm2_b  = (const float*)d_in[16];

    const int M = 4096;   // B*S tokens
    char* cur = (char*)d_ws;
    auto alloc = [&](size_t bytes) -> void* {
        void* p = (void*)cur;
        cur += (bytes + 255) & ~(size_t)255;
        return p;
    };
    u16* qkv_wT = (u16*)alloc((size_t)3072 * 1024 * 2);
    u16* out_wT = (u16*)alloc((size_t)1024 * 1024 * 2);
    u16* w1T    = (u16*)alloc((size_t)4096 * 1024 * 2);
    u16* w2T    = (u16*)alloc((size_t)1024 * 4096 * 2);
    u16* h_bf   = (u16*)alloc((size_t)M * 1024 * 2);
    u16* qkv_bf = (u16*)alloc((size_t)M * 3072 * 2);
    u16* ctx_bf = (u16*)alloc((size_t)M * 1024 * 2);
    float* y1   = (float*)alloc((size_t)M * 1024 * 4);
    float* x1   = (float*)alloc((size_t)M * 1024 * 4);
    u16* h2_bf  = (u16*)alloc((size_t)M * 1024 * 2);
    u16* ff1_bf = (u16*)alloc((size_t)M * 4096 * 2);
    float* y2   = (float*)alloc((size_t)M * 1024 * 4);
    (void)in_sizes; (void)n_in; (void)out_size; (void)ws_size;

    // weight transpose-converts
    convT_kernel<<<(1024 * 3072) / 256, 256, 0, stream>>>(qkv_w, qkv_wT, 1024, 3072);
    convT_kernel<<<(1024 * 1024) / 256, 256, 0, stream>>>(out_w, out_wT, 1024, 1024);
    convT_kernel<<<(1024 * 4096) / 256, 256, 0, stream>>>(w1, w1T, 1024, 4096);
    convT_kernel<<<(4096 * 1024) / 256, 256, 0, stream>>>(w2, w2T, 4096, 1024);

    // h = LN(x, attn_ln) -> bf16
    ln_kernel<<<M, 256, 0, stream>>>(x, nullptr, attn_ln_g, attn_ln_b, nullptr, h_bf);
    // qkv = h @ qkv_w + qkv_b -> bf16
    gemm_bf16_kernel<<<dim3(3072 / 128, M / 128), 256, 0, stream>>>(
        h_bf, qkv_wT, qkv_b, nullptr, nullptr, qkv_bf, M, 3072, 1024, 0);
    // attention -> ctx bf16 [B,S,D]
    flash_attn_kernel<<<dim3(16, 16, 2), 256, 0, stream>>>(qkv_bf, ctx_bf);
    // y1 = x + ctx @ out_w + out_b
    gemm_bf16_kernel<<<dim3(1024 / 128, M / 128), 256, 0, stream>>>(
        ctx_bf, out_wT, out_b, x, y1, nullptr, M, 1024, 1024, 0);
    // x1 = LN(y1, norm1)
    ln_kernel<<<M, 256, 0, stream>>>(y1, nullptr, norm1_g, norm1_b, x1, nullptr);
    // h2 = LN(x1, ff_ln) -> bf16
    ln_kernel<<<M, 256, 0, stream>>>(x1, nullptr, ff_ln_g, ff_ln_b, nullptr, h2_bf);
    // ff1 = gelu(h2 @ w1 + b1) -> bf16
    gemm_bf16_kernel<<<dim3(4096 / 128, M / 128), 256, 0, stream>>>(
        h2_bf, w1T, b1, nullptr, nullptr, ff1_bf, M, 4096, 1024, 1);
    // y2 = x1 + ff1 @ w2 + b2
    gemm_bf16_kernel<<<dim3(1024 / 128, M / 128), 256, 0, stream>>>(
        ff1_bf, w2T, b2, x1, y2, nullptr, M, 1024, 4096, 0);
    // out = LN(y2, norm2)
    ln_kernel<<<M, 256, 0, stream>>>(y2, nullptr, norm2_g, norm2_b, (float*)d_out, nullptr);
}
